// GCNScorer_64707977281657
// MI455X (gfx1250) — compile-verified, matrix operand order checked by reading, not run on hardware
//
#include <hip/hip_runtime.h>
#include <hip/hip_fp16.h>
#include <stdint.h>

typedef __attribute__((ext_vector_type(16))) _Float16 v16h;
typedef __attribute__((ext_vector_type(8)))  float    v8f;

#define NNODES 100000   // multiple of 16 -> no tail handling in WMMA tiles
#define EEDGES 1600000
#define NCAND  1000000

// K-index mapping for 16-bit A/B fragments of v_wmma_f32_16x16x32_f16 (wave32):
// lanes 0-15 hold K 0..7 (V0..V3) and 16..23 (V4..V7); lanes 16-31 hold K 8..15 and 24..31.
__device__ __forceinline__ int frag_k(int e, int half) {
    return (e < 8) ? (8 * half + e) : (16 + 8 * half + (e - 8));
}

// ---------------- degree / normalization ----------------
__global__ void k_deg_init(float* __restrict__ deg) {
    int i = blockIdx.x * blockDim.x + threadIdx.x;
    if (i < NNODES) deg[i] = 1.0f;               // self-loop contribution
}

__global__ void k_deg_accum(const long long* __restrict__ col, float* __restrict__ deg) {
    int e = blockIdx.x * blockDim.x + threadIdx.x;
    if (e < EEDGES) unsafeAtomicAdd(&deg[(int)col[e]], 1.0f);  // hw global_atomic_add_f32
}

__global__ void k_rsqrt(float* __restrict__ deg) {
    int i = blockIdx.x * blockDim.x + threadIdx.x;
    if (i < NNODES) deg[i] = rsqrtf(deg[i]);     // deg >= 1 always (self-loops)
}

// ---------------- dense feature transform via WMMA, fused epilogue ----------------
// h   = (RELU_IN ? relu(in) : in) @ W                       [N, KOUT]
// agg = bias + dis^2 * h        (self-loop message + bias)  [N, KOUT]
template <int KIN, int KOUT, bool RELU_IN>
__global__ void k_transform_wmma(const float* __restrict__ in,
                                 const float* __restrict__ W,
                                 const float* __restrict__ bias,
                                 const float* __restrict__ dis,
                                 float* __restrict__ out_h,
                                 float* __restrict__ out_agg,
                                 int n) {
    constexpr int KSTEPS = (KIN + 31) / 32;
    constexpr int CT     = KOUT / 16;

    const int lane = threadIdx.x & 31;
    const int wave = threadIdx.x >> 5;
    const int tile = blockIdx.x * (blockDim.x >> 5) + wave;
    const int nTiles = n >> 4;                   // n % 16 == 0
    if (tile >= nTiles) return;                  // uniform per wave -> EXEC stays all-ones

    const int half = lane >> 4;   // which 16-lane half
    const int mn   = lane & 15;   // row (A) / col (B,C) within tile

    // B fragments once per wave (W is tiny, L2-resident).
    v16h bfrag[CT * KSTEPS];
#pragma unroll
    for (int ct = 0; ct < CT; ++ct)
#pragma unroll
        for (int ks = 0; ks < KSTEPS; ++ks) {
            v16h b;
#pragma unroll
            for (int e = 0; e < 16; ++e) {
                int k = ks * 32 + frag_k(e, half);
                float v = (k < KIN) ? W[k * KOUT + ct * 16 + mn] : 0.0f;
                b[e] = (_Float16)v;
            }
            bfrag[ct * KSTEPS + ks] = b;
        }

    v8f acc[CT] = {};

    const int row0 = tile * 16;
    const float* rp = in + (size_t)(row0 + mn) * KIN;

#pragma unroll
    for (int ks = 0; ks < KSTEPS; ++ks) {
        v16h a;
#pragma unroll
        for (int e = 0; e < 16; ++e) {
            int k = ks * 32 + frag_k(e, half);
            float v = (k < KIN) ? rp[k] : 0.0f;
            if (RELU_IN) v = fmaxf(v, 0.0f);     // fused relu of layer-1 aggregate
            a[e] = (_Float16)v;
        }
#pragma unroll
        for (int ct = 0; ct < CT; ++ct)
            acc[ct] = __builtin_amdgcn_wmma_f32_16x16x32_f16(
                false, a, false, bfrag[ct * KSTEPS + ks], (short)0, acc[ct], false, false);
    }

    // C/D layout: vgpr rr -> (M = 8*half + rr, N = mn). No per-element guards:
    // rows are always < n. Address as base + rr*KOUT so offsets fold into the store.
    const int rbase = row0 + half * 8;
    float d2[8];
#pragma unroll
    for (int rr = 0; rr < 8; ++rr) {
        float d = dis[rbase + rr];
        d2[rr] = d * d;                          // self-loop norm = dis[v]^2
    }
#pragma unroll
    for (int ct = 0; ct < CT; ++ct) {
        const float bv = bias[ct * 16 + mn];
        float* ph = out_h   + (size_t)rbase * KOUT + ct * 16 + mn;
        float* pa = out_agg + (size_t)rbase * KOUT + ct * 16 + mn;
#pragma unroll
        for (int rr = 0; rr < 8; ++rr) {
            float v = acc[ct][rr];
            ph[(size_t)rr * KOUT] = v;
            pa[(size_t)rr * KOUT] = bv + d2[rr] * v;
        }
    }
}

// ---------------- edge scatter (message passing) ----------------
// One wave per edge; 64 feats -> 2 per lane (two coalesced 128B halves).
__global__ void k_scatter64(const long long* __restrict__ srcIdx, const long long* __restrict__ dstIdx,
                            const float* __restrict__ dis, const float* __restrict__ h,
                            float* __restrict__ agg) {
    int e    = (blockIdx.x * blockDim.x + threadIdx.x) >> 5;
    int lane = threadIdx.x & 31;
    if (e >= EEDGES) return;
    __builtin_prefetch(srcIdx + e + 2048, 0, 0);   // global_prefetch_b8, streaming indices
    __builtin_prefetch(dstIdx + e + 2048, 0, 0);
    int s = (int)srcIdx[e];
    int t = (int)dstIdx[e];
    float w = dis[s] * dis[t];
    const float* hs = h + (size_t)s * 64;
    float* at = agg + (size_t)t * 64;
    unsafeAtomicAdd(&at[lane],      w * hs[lane]);
    unsafeAtomicAdd(&at[lane + 32], w * hs[lane + 32]);
}

// One wave per edge; 32 feats == wave32 lanes exactly.
__global__ void k_scatter32(const long long* __restrict__ srcIdx, const long long* __restrict__ dstIdx,
                            const float* __restrict__ dis, const float* __restrict__ h,
                            float* __restrict__ agg) {
    int e    = (blockIdx.x * blockDim.x + threadIdx.x) >> 5;
    int lane = threadIdx.x & 31;
    if (e >= EEDGES) return;
    __builtin_prefetch(srcIdx + e + 2048, 0, 0);
    __builtin_prefetch(dstIdx + e + 2048, 0, 0);
    int s = (int)srcIdx[e];
    int t = (int)dstIdx[e];
    float w = dis[s] * dis[t];
    unsafeAtomicAdd(&agg[(size_t)t * 32 + lane], w * h[(size_t)s * 32 + lane]);
}

// ---------------- candidate scoring ----------------
// One wave per candidate: coalesced 128B gather of each endpoint's 32 feats,
// weighted by Ws, wave-shuffle reduction, sigmoid on lane 0.
__global__ void k_score(const long long* __restrict__ cand, const float* __restrict__ h2,
                        const float* __restrict__ Ws, const float* __restrict__ bs,
                        float* __restrict__ out) {
    int c    = (blockIdx.x * blockDim.x + threadIdx.x) >> 5;
    int lane = threadIdx.x & 31;
    if (c >= NCAND) return;
    long long u = cand[(size_t)c * 2 + 0];
    long long v = cand[(size_t)c * 2 + 1];
    float p = h2[(size_t)u * 32 + lane] * Ws[lane]
            + h2[(size_t)v * 32 + lane] * Ws[32 + lane];
#pragma unroll
    for (int off = 16; off; off >>= 1) p += __shfl_xor(p, off, 32);
    if (lane == 0) out[c] = 1.0f / (1.0f + __expf(-(p + bs[0])));
}

// ---------------- launch ----------------
extern "C" void kernel_launch(void* const* d_in, const int* in_sizes, int n_in,
                              void* d_out, int out_size, void* d_ws, size_t ws_size,
                              hipStream_t stream) {
    const float*     x    = (const float*)d_in[0];
    const long long* ei   = (const long long*)d_in[1];   // [2,E]: src row then dst row
    const long long* cand = (const long long*)d_in[2];   // [C,2]
    const float*     W1   = (const float*)d_in[3];
    const float*     b1   = (const float*)d_in[4];
    const float*     W2   = (const float*)d_in[5];
    const float*     b2   = (const float*)d_in[6];
    const float*     Ws   = (const float*)d_in[7];
    const float*     bs   = (const float*)d_in[8];
    float*           out  = (float*)d_out;

    const long long* srcIdx = ei;            // edge_index[0]
    const long long* dstIdx = ei + EEDGES;   // edge_index[1]

    // Workspace layout (floats). hx (N*64) is dead after layer-1 scatter,
    // so layer-2 buffers (N*32 + N*32) overlay it.  Total ~52 MB.
    float* ws   = (float*)d_ws;
    float* dis  = ws;                               // [N]       deg -> rsqrt(deg)
    float* hx   = dis + NNODES;                     // [N,64]    x @ W1
    float* agg1 = hx + (size_t)NNODES * 64;         // [N,64]    layer-1 aggregate (pre-relu)
    float* hw2  = hx;                               // [N,32]    relu(h1) @ W2 (reuses hx)
    float* agg2 = hx + (size_t)NNODES * 32;         // [N,32]    layer-2 aggregate -> h2

    const int T = 256;
    const int nTiles     = NNODES / 16;                 // 6250
    const int gTiles     = (nTiles + 7) / 8;            // 8 waves/block
    const int gEdgeWaves = (EEDGES + 7) / 8;            // 1 wave/edge
    const int gCandWaves = (NCAND + 7) / 8;             // 1 wave/candidate

    // normalization
    k_deg_init <<<(NNODES + T - 1) / T, T, 0, stream>>>(dis);
    k_deg_accum<<<(EEDGES + T - 1) / T, T, 0, stream>>>(dstIdx, dis);
    k_rsqrt    <<<(NNODES + T - 1) / T, T, 0, stream>>>(dis);

    // layer 1: WMMA transform with fused (bias + self-loop) epilogue, then edge scatter
    k_transform_wmma<7, 64, false><<<gTiles, T, 0, stream>>>(x, W1, b1, dis, hx, agg1, NNODES);
    k_scatter64<<<gEdgeWaves, T, 0, stream>>>(srcIdx, dstIdx, dis, hx, agg1);

    // layer 2: WMMA transform (relu fused on load) with fused epilogue, then edge scatter
    k_transform_wmma<64, 32, true><<<gTiles, T, 0, stream>>>(agg1, W2, b2, dis, hw2, agg2, NNODES);
    k_scatter32<<<gEdgeWaves, T, 0, stream>>>(srcIdx, dstIdx, dis, hw2, agg2);

    // scoring
    k_score<<<gCandWaves, T, 0, stream>>>(cand, agg2, Ws, bs, out);
}
